// VFXNetPatchDecoder_28011776704577
// MI455X (gfx1250) — matrix operand order repaired and checked by reading, not compiled, not run on hardware
//
#include <hip/hip_runtime.h>

// ---------------------------------------------------------------------------
// Types (trivial ext-vectors so address-space-qualified loads are legal)
// ---------------------------------------------------------------------------
typedef __attribute__((ext_vector_type(16))) _Float16     v16h;
typedef __attribute__((ext_vector_type(8)))  float        v8f;
typedef __attribute__((ext_vector_type(4)))  _Float16     h4;
typedef __attribute__((ext_vector_type(4)))  unsigned int u32x4;
typedef __attribute__((ext_vector_type(2)))  int          i32x2;
typedef __attribute__((ext_vector_type(2)))  float        f32x2;
typedef __attribute__((ext_vector_type(4)))  float        f32x4;

// Explicit address spaces: 1 = global, 3 = LDS -> force global_* / ds_* ops
#define AS1 __attribute__((address_space(1)))
#define AS3 __attribute__((address_space(3)))
typedef AS3 _Float16       lf16;    // LDS f16
typedef AS3 h4             lh4;     // LDS 4xf16 (b64 store)
typedef AS3 u32x4          lu32x4;  // LDS 16B (ds_load_b128)
typedef AS1 const _Float16 gf16;    // global f16 (weights)
typedef AS1 const u32x4    gu32x4;  // global 16B (global_load_b128)

#define HW 2048            // latent H == W
#define RS 136             // activation row stride in f16 (272B -> conflict-free b128)
#define ACT_ELEMS (16*RS)  // one 16-row activation buffer

// Weight image layout in d_ws (f16 units):
//   L1: 16 frags @ 0      (K 108->128 pad, N=64 : nt=4, ks=4)
//   L2: 16 frags @ 8192   (K=64,  N=128     : nt=8, ks=2)
//   L3: 16 frags @ 16384  (K=128, N=64      : nt=4, ks=4)
//   L4:  2 frags @ 24576  (K=64,  N 4->16   : nt=1, ks=2)
// Each fragment = 512 f16 (32 lanes x 16 f16, lane-contiguous), 1KB.
// Biases (f32, 272 = 64+128+64+16) at byte offset 51200.

union F16Frag { u32x4 u[2]; v16h h; };

__device__ inline v16h load_a_frag(const lf16* actIn, int m, int hh, int ks) {
  // 16-bit A 16x32 layout: lane<16 -> M=lane, K = [0,8)+[16,24);
  //                        lane>=16 -> M=lane-16, K = [8,16)+[24,32)
  F16Frag f;
  const lf16* p = actIn + m * RS + ks * 32 + hh * 8;
  f.u[0] = *(const lu32x4*)(p);        // ds_load_b128
  f.u[1] = *(const lu32x4*)(p + 16);   // ds_load_b128
  return f.h;
}

__device__ inline v16h load_b_frag(const gf16* frag, int l) {
  F16Frag f;
  const gu32x4* p = (const gu32x4*)(frag + l * 16);
  f.u[0] = p[0];                       // global_load_b128
  f.u[1] = p[1];                       // global_load_b128
  return f.h;
}

// Branch-free tanh-form GELU: 1 v_exp_f32 + 1 v_rcp_f32 + a few FMAs
__device__ inline float gelu_fast(float x) {
  const float u = 0.7978845608028654f * x * (1.0f + 0.044715f * x * x);
  const float e = __expf(2.0f * u);                                // v_exp_f32
  const float t = 1.0f - 2.0f * __builtin_amdgcn_rcpf(e + 1.0f);   // tanh(u)
  return 0.5f * x * (1.0f + t);
}

// Hidden layer with B-matrix fragments resident in VGPRs
template <int KSTEPS, int NTILES>
__device__ inline void mlp_layer_reg(const lf16* aIn, lf16* aOut,
                                     const v16h* wB, const float* bias,
                                     int m, int hh) {
  v16h a[KSTEPS];
#pragma unroll
  for (int ks = 0; ks < KSTEPS; ++ks) a[ks] = load_a_frag(aIn, m, hh, ks);
#pragma unroll
  for (int nt = 0; nt < NTILES; ++nt) {
    v8f acc = {};
#pragma unroll
    for (int ks = 0; ks < KSTEPS; ++ks) {
      acc = __builtin_amdgcn_wmma_f32_16x16x32_f16(
          false, a[ks], false, wB[nt * KSTEPS + ks], (short)0, acc, false, false);
    }
    const float bv = bias[nt];
#pragma unroll
    for (int v = 0; v < 8; ++v)   // D: VGPR v, lane l -> M=v+8*hh, N=m
      aOut[(v + 8 * hh) * RS + nt * 16 + m] = (_Float16)gelu_fast(acc[v] + bv);
  }
}

// Hidden layer with B-matrix fragments streamed from global (WGP$-resident)
template <int KSTEPS, int NTILES>
__device__ inline void mlp_layer_gbl(const lf16* aIn, lf16* aOut,
                                     const gf16* wBase, const float* bias,
                                     int l, int m, int hh) {
  v16h a[KSTEPS];
#pragma unroll
  for (int ks = 0; ks < KSTEPS; ++ks) a[ks] = load_a_frag(aIn, m, hh, ks);
#pragma unroll
  for (int nt = 0; nt < NTILES; ++nt) {
    v8f acc = {};
#pragma unroll
    for (int ks = 0; ks < KSTEPS; ++ks) {
      v16h b = load_b_frag(wBase + (nt * KSTEPS + ks) * 512, l);
      acc = __builtin_amdgcn_wmma_f32_16x16x32_f16(
          false, a[ks], false, b, (short)0, acc, false, false);
    }
    const float bv = bias[nt];
#pragma unroll
    for (int v = 0; v < 8; ++v)
      aOut[(v + 8 * hh) * RS + nt * 16 + m] = (_Float16)gelu_fast(acc[v] + bv);
  }
}

// Issue the 9-tap latent gather for one tile (this lane's 4-5 taps).
__device__ inline void issue_gather(const AS1 f32x4* lat4, i32x2 xy, int hh,
                                    f32x4 lv[5]) {
  const int px = xy.x, py = xy.y;
#pragma unroll
  for (int i = 0; i < 5; ++i) {
    const int tap = 2 * i + hh;
    if (tap <= 8) {  // compile-time true for i<4; runtime (hh==0) for i==4
      const int dx = tap % 3 - 1, dy = tap / 3 - 1;
      const int ex = min(max(px + dx, 0), HW - 1);
      const int ey = min(max(py + dy, 0), HW - 1);
      lv[i] = lat4[ey * HW + ex];      // global_load_b128 (L2-resident)
    }
  }
}

// Convert + store one tile's 16x108 input patch into P (f16), plus K-pad zeros.
__device__ inline void write_input(lf16* P, const f32x4 lv[5], i32x2 xy, f32x2 cc,
                                   int m, int hh, bool rezero) {
  const int px = xy.x, py = xy.y;
  const float xs = (float)min(max(px, 0), HW - 1) * (1.0f / HW);
  const float ys = (float)min(max(py, 0), HW - 1) * (1.0f / HW);
  const float tp = 6.28318530717958648f;
  const float sx = __sinf(tp * xs), cx = __cosf(tp * xs);
  const float sy = __sinf(tp * ys), cy = __cosf(tp * ys);
  const h4 pc0 = { (_Float16)xs, (_Float16)ys, (_Float16)sx, (_Float16)cx };
  const h4 pc1 = { (_Float16)sy, (_Float16)cy, (_Float16)cc.x, (_Float16)cc.y };

  lf16* row = P + m * RS;
#pragma unroll
  for (int i = 0; i < 5; ++i) {
    const int tap = 2 * i + hh;      // hh=0: taps 0,2,4,6,8 ; hh=1: taps 1,3,5,7
    if (tap <= 8) {
      const f32x4 v = lv[i];
      const h4 lh = { (_Float16)v.x, (_Float16)v.y, (_Float16)v.z, (_Float16)v.w };
      *(lh4*)(row + tap * 12)     = lh;   // ds_store_b64, 8B aligned
      *(lh4*)(row + tap * 12 + 4) = pc0;
      *(lh4*)(row + tap * 12 + 8) = pc1;
    }
  }
  if (rezero) {
    // L2 wrote cols 0..127 of P this iteration: re-zero K-pad cols 108..127.
    const h4 z = {};
#pragma unroll
    for (int j = 0; j < 5; ++j)
      if ((j & 1) == hh) *(lh4*)(row + 108 + j * 4) = z;
  }
}

// ---------------------------------------------------------------------------
// One-time weight repack: f32 row-major (K,N) -> f16 WMMA B-fragment order
// ---------------------------------------------------------------------------
__global__ __launch_bounds__(256) void vfx_prep_kernel(
    const float* __restrict__ W1, const float* __restrict__ b1,
    const float* __restrict__ W2, const float* __restrict__ b2,
    const float* __restrict__ W3, const float* __restrict__ b3,
    const float* __restrict__ W4, const float* __restrict__ b4,
    _Float16* __restrict__ wsW, float* __restrict__ wsB) {
  int e = blockIdx.x * blockDim.x + threadIdx.x;
  if (e < 25600) {
    const float* Wp; int Kr, Nr, KS; int rem = e;
    if (rem < 8192)       { Wp = W1; Kr = 108; Nr = 64;  KS = 4; }
    else if (rem < 16384) { Wp = W2; Kr = 64;  Nr = 128; KS = 2; rem -= 8192;  }
    else if (rem < 24576) { Wp = W3; Kr = 128; Nr = 64;  KS = 4; rem -= 16384; }
    else                  { Wp = W4; Kr = 64;  Nr = 4;   KS = 2; rem -= 24576; }
    int frag = rem >> 9;       // fragment index = nt*KS + ks
    int r    = rem & 511;
    int lane = r >> 4;
    int slot = r & 15;
    int nt = frag / KS;
    int ks = frag - nt * KS;
    // B 32x16 layout: lane<16 -> N=lane,     K = ks*32 + [0,8)+[16,24)
    //                 lane>=16 -> N=lane-16, K = ks*32 + [8,16)+[24,32)
    int k = ks * 32 + ((lane >= 16) ? 8 : 0) + slot + ((slot >= 8) ? 8 : 0);
    int n = nt * 16 + (lane & 15);
    float w = (k < Kr && n < Nr) ? Wp[k * Nr + n] : 0.0f;
    wsW[e] = (_Float16)w;
  } else if (e < 25872) {
    int i = e - 25600;
    float v;
    if (i < 64)       v = b1[i];
    else if (i < 192) v = b2[i - 64];
    else if (i < 256) v = b3[i - 192];
    else              v = (i - 256 < 4) ? b4[i - 256] : 0.0f;
    wsB[i] = v;
  }
}

// ---------------------------------------------------------------------------
// Main decoder: persistent waves, 16 points per wave-iteration, WMMA MLP.
// L2/L3/L4 weights pinned in VGPRs; L1 streamed from WGP$; latent gather
// software-pipelined one iteration ahead (store-at-end into actA).
// ---------------------------------------------------------------------------
__global__ __launch_bounds__(128, 2) void vfx_decoder_kernel(
    const float* __restrict__ latent_, const int* __restrict__ raw_pos_,
    const float* __restrict__ control_, const _Float16* __restrict__ wsW_,
    const float* __restrict__ wsB_, float* __restrict__ out_,
    int N, int numTiles) {
  __shared__ __align__(16) _Float16 act_[4 * 2 * ACT_ELEMS];  // 4 waves x 2 bufs

  lf16*            act  = (lf16*)act_;
  const AS1 f32x4* lat4 = (const AS1 f32x4*)latent_;   // latent is [H][W][4] f32
  const AS1 i32x2* rpos = (const AS1 i32x2*)raw_pos_;
  const AS1 f32x2* ctrl = (const AS1 f32x2*)control_;
  const gf16*      wsW  = (const gf16*)wsW_;
  const AS1 float* wsB  = (const AS1 float*)wsB_;
  AS1 float*       out  = (AS1 float*)out_;

  const int wave = threadIdx.x >> 5;
  const int l  = threadIdx.x & 31;
  const int hh = l >> 4;   // half-wave (WMMA lane group)
  const int m  = l & 15;   // point row within tile / matrix N index

  lf16* actA = act + wave * (2 * ACT_ELEMS);  // input + L2 output
  lf16* actB = actA + ACT_ELEMS;              // L1/L3 output

  // Per-lane biases in registers (index depends only on m)
  float bias1[4], bias2[8], bias3[4];
#pragma unroll
  for (int nt = 0; nt < 4; ++nt) bias1[nt] = wsB[nt * 16 + m];
#pragma unroll
  for (int nt = 0; nt < 8; ++nt) bias2[nt] = wsB[64 + nt * 16 + m];
#pragma unroll
  for (int nt = 0; nt < 4; ++nt) bias3[nt] = wsB[192 + nt * 16 + m];
  const float bias4 = wsB[256 + m];

  // Pin L2/L3/L4 weights (34 frags = 272 VGPRs) in registers.
  v16h wB2[16], wB3[16], wB4[2];
#pragma unroll
  for (int i = 0; i < 16; ++i) wB2[i] = load_b_frag(wsW + 8192  + i * 512, l);
#pragma unroll
  for (int i = 0; i < 16; ++i) wB3[i] = load_b_frag(wsW + 16384 + i * 512, l);
#pragma unroll
  for (int i = 0; i < 2;  ++i) wB4[i] = load_b_frag(wsW + 24576 + i * 512, l);

  // Zero cols 108..135 of actA once (cols 128..135 are never written again;
  // cols 108..127 are re-zeroed each iteration in write_input).
  {
    const h4 z = {};
    lf16* row = actA + m * RS;
#pragma unroll
    for (int j = 0; j < 7; ++j)
      if ((j & 1) == hh) *(lh4*)(row + 108 + j * 4) = z;
  }

  const int stride = gridDim.x * 4;
  int t = blockIdx.x * 4 + wave;
  if (t >= numTiles) return;

  // ---- pipeline prologue: tile t's input into actA; coords for t+stride ----
  int p0 = t * 16 + m;               if (p0 >= N) p0 = N - 1;
  int p1 = (t + stride) * 16 + m;    if (p1 >= N) p1 = N - 1;
  {
    const i32x2 xy0 = rpos[p0];
    const f32x2 cc0 = ctrl[p0];
    f32x4 lv0[5];
    issue_gather(lat4, xy0, hh, lv0);
    write_input(actA, lv0, xy0, cc0, m, hh, /*rezero=*/false);
  }
  i32x2 xyNxt = rpos[p1];
  f32x2 ccNxt = ctrl[p1];

  int w1off = 0;  // laundered each iteration: keeps L1 weight loads in-loop
  for (; t < numTiles; t += stride) {
    const int base = t * 16;

    // ---- 1. prefetch: latent gathers for t+stride, coords for t+2*stride ----
    f32x4 lvN[5];
    issue_gather(lat4, xyNxt, hh, lvN);
    int p2 = (t + 2 * stride) * 16 + m;  if (p2 >= N) p2 = N - 1;
    const i32x2 xy2 = rpos[p2];
    const f32x2 cc2 = ctrl[p2];

    // ---- 2. MLP: A=16x108 @ 108x64 -> 16x128 -> 16x64 -> 16x4 ----
    asm volatile("" : "+s"(w1off));  // defeat LICM: L1 B-frags reload from WGP$
    mlp_layer_gbl<4, 4>(actA, actB, wsW + w1off, bias1, l, m, hh);
    mlp_layer_reg<2, 8>(actB, actA, wB2, bias2, m, hh);   // writes actA 0..127
    mlp_layer_reg<4, 4>(actA, actB, wB3, bias3, m, hh);

    // final layer: K=64, N padded 4->16, sigmoid, write f32 out
    {
      v16h a0 = load_a_frag(actB, m, hh, 0);
      v16h a1 = load_a_frag(actB, m, hh, 1);
      v8f acc = {};
      acc = __builtin_amdgcn_wmma_f32_16x16x32_f16(false, a0, false, wB4[0], (short)0, acc, false, false);
      acc = __builtin_amdgcn_wmma_f32_16x16x32_f16(false, a1, false, wB4[1], (short)0, acc, false, false);
      if (m < 4) {
#pragma unroll
        for (int v = 0; v < 8; ++v) {
          const int gp = base + v + 8 * hh;
          if (gp < N) {
            const float x = acc[v] + bias4;
            out[gp * 4 + m] = __builtin_amdgcn_rcpf(1.0f + __expf(-x));
          }
        }
      }
    }

    // ---- 3. store next tile's input into actA (DS in-order within wave:
    //         L1/L3 loads above read pre-store data) ----
    write_input(actA, lvN, xyNxt, ccNxt, m, hh, /*rezero=*/true);

    // ---- 4. rotate coords ----
    xyNxt = xy2;
    ccNxt = cc2;
  }
}

// ---------------------------------------------------------------------------
// Host launch
// ---------------------------------------------------------------------------
extern "C" void kernel_launch(void* const* d_in, const int* in_sizes, int n_in,
                              void* d_out, int out_size, void* d_ws, size_t ws_size,
                              hipStream_t stream) {
  (void)n_in; (void)out_size; (void)ws_size;
  const float* latent  = (const float*)d_in[0];
  const int*   raw_pos = (const int*)d_in[1];
  const float* control = (const float*)d_in[2];
  const float* W1 = (const float*)d_in[3];
  const float* b1 = (const float*)d_in[4];
  const float* W2 = (const float*)d_in[5];
  const float* b2 = (const float*)d_in[6];
  const float* W3 = (const float*)d_in[7];
  const float* b3 = (const float*)d_in[8];
  const float* W4 = (const float*)d_in[9];
  const float* b4 = (const float*)d_in[10];
  float* out = (float*)d_out;

  const int N = in_sizes[1] / 2;  // raw_pos is [N,2]
  _Float16* wsW = (_Float16*)d_ws;
  float*    wsB = (float*)((char*)d_ws + 25600 * 2);

  vfx_prep_kernel<<<(25872 + 255) / 256, 256, 0, stream>>>(
      W1, b1, W2, b2, W3, b3, W4, b4, wsW, wsB);

  const int numTiles = (N + 15) / 16;
  const int blocks = 2048;  // 8192 persistent waves, ~7.6 pipelined iters each
  vfx_decoder_kernel<<<blocks, 128, 0, stream>>>(
      latent, raw_pos, control, wsW, wsB, out, N, numTiles);
}
